// Transformer_53257594471038
// MI455X (gfx1250) — matrix operand order repaired and checked by reading, not compiled, not run on hardware
//
#include <hip/hip_runtime.h>
#include <stdint.h>

// ---------------- problem constants ----------------
#define B_      64
#define N_      196
#define DIM_    768
#define EMBED_  384
#define CDIM_   384
#define HEADS_  8
#define DH_     64
#define INNER_  512          // HEADS*DH
#define VD_     896
#define DV_     112
#define MROWS_  (B_ * N_)    // 12544 = 98 * 128
#define NPAD_   208          // 13*16 padded token rows for q/k
#define KPAD_   224          // 14*16 padded key dim for v^T
#define EPS_    1e-5f

typedef __attribute__((ext_vector_type(16))) __bf16 v16bf;
typedef __attribute__((ext_vector_type(8)))  float  v8f;

// ---------------- helpers ----------------
__device__ __forceinline__ unsigned short f2bf(float f) {
  union { float f; uint32_t u; } x; x.f = f;
  uint32_t u = x.u;
  uint32_t r = u + 0x7FFFu + ((u >> 16) & 1u);   // round-to-nearest-even
  return (unsigned short)(r >> 16);
}

__device__ __forceinline__ v8f wmma_bf16(v16bf a, v16bf b, v8f c) {
  return __builtin_amdgcn_wmma_f32_16x16x32_bf16(false, a, false, b, (short)0, c,
                                                 false, false);
}

// A-fragment (16x32 bf16, M on lanes, interleaved K-pair map per ISA 7.12.2):
// lane = 16g+m ; dword i<4 -> Kpair i+4g ; dword 4+i -> Kpair 8+i+4g
__device__ __forceinline__ v16bf load_frag_a(const unsigned short* base, int ld,
                                             int rowclamp) {
  int lane = threadIdx.x & 31;
  int g = lane >> 4, m = lane & 15;
  int r = m > rowclamp ? rowclamp : m;
  if (r < 0) r = 0;
  const uint32_t* p = (const uint32_t*)(base + (size_t)r * ld);
  union { v16bf v; uint32_t u[8]; } t;
#pragma unroll
  for (int i = 0; i < 4; ++i) { t.u[i] = p[i + 4 * g]; t.u[4 + i] = p[8 + i + 4 * g]; }
  return t.v;
}

// B-fragment (32x16 bf16, N on lanes, interleaved K-pair map matching A):
// source is K-contiguous per output column (i.e. transposed weight row).
// lane = 16g+n ; dword i<4 -> Kpair i+4g ; dword 4+i -> Kpair 8+i+4g
__device__ __forceinline__ v16bf load_frag_b(const unsigned short* base, int ld,
                                             int rowclamp) {
  int lane = threadIdx.x & 31;
  int g = lane >> 4, n = lane & 15;
  int r = n > rowclamp ? rowclamp : n;
  if (r < 0) r = 0;
  const uint32_t* p = (const uint32_t*)(base + (size_t)r * ld);
  union { v16bf v; uint32_t u[8]; } t;
#pragma unroll
  for (int i = 0; i < 4; ++i) { t.u[i] = p[i + 4 * g]; t.u[4 + i] = p[8 + i + 4 * g]; }
  return t.v;
}

// ---------------- BN stats: per-channel scale/shift ----------------
__global__ __launch_bounds__(256)
void bn_stats_kernel(const float* __restrict__ x, const float* __restrict__ gam,
                     const float* __restrict__ bet, float* __restrict__ scl,
                     float* __restrict__ shf, int rows, int C) {
  int c = blockIdx.x;
  float s = 0.f, s2 = 0.f;
  for (int i = threadIdx.x; i < rows; i += blockDim.x) {
    float v = x[(size_t)i * C + c];
    s += v; s2 += v * v;
  }
  __shared__ float r1[256], r2[256];
  int tid = threadIdx.x;
  r1[tid] = s; r2[tid] = s2;
  __syncthreads();
  for (int off = 128; off > 0; off >>= 1) {
    if (tid < off) { r1[tid] += r1[tid + off]; r2[tid] += r2[tid + off]; }
    __syncthreads();
  }
  if (tid == 0) {
    float inv = 1.0f / (float)rows;
    float mean = r1[0] * inv;
    float var  = r2[0] * inv - mean * mean;   // biased, like torch BN
    float rs   = rsqrtf(var + EPS_);
    float sc   = gam[c] * rs;
    scl[c] = sc;
    shf[c] = bet[c] - mean * sc;
  }
}

__global__ __launch_bounds__(256)
void norm_kernel(const float* __restrict__ x, const float* __restrict__ scl,
                 const float* __restrict__ shf, unsigned short* __restrict__ out,
                 int total, int C) {
  int i = blockIdx.x * blockDim.x + threadIdx.x;
  if (i >= total) return;
  int c = i % C;
  out[i] = f2bf(x[i] * scl[c] + shf[c]);
}

__global__ __launch_bounds__(256)
void gelu_kernel(const float* __restrict__ f, const float* __restrict__ scl,
                 const float* __restrict__ shf, unsigned short* __restrict__ out,
                 int total, int C) {
  int i = blockIdx.x * blockDim.x + threadIdx.x;
  if (i >= total) return;
  int c = i % C;
  float y = f[i] * scl[c] + shf[c];
  float g = 0.5f * y * (1.0f + erff(y * 0.70710678118654752f));  // exact GELU
  out[i] = f2bf(g);
}

// transpose + f32->bf16: W (KxN row-major) -> WT (NxK row-major)
__global__ __launch_bounds__(256)
void wtrans_kernel(const float* __restrict__ W, unsigned short* __restrict__ WT,
                   int K, int N) {
  int i = blockIdx.x * blockDim.x + threadIdx.x;
  if (i >= K * N) return;
  int nrow = i / K, k = i % K;
  WT[i] = f2bf(W[(size_t)k * N + nrow]);
}

// ---------------- generic bf16 WMMA GEMM ----------------
// C[M x N] = A[M x K](bf16) * BT[N x K]^T(bf16) + bias.
// Block tile 128x128, 8 wave32 waves, wave tile 32x64 (8 WMMAs per K-step).
// Double-buffered LDS staging; global loads for step k+1 issued before the
// WMMA block of step k; prefetch of step k+2 tiles (global_prefetch_b8).
// Epilogue by EPI:
//   0: scatter into q / k buffers ((b,h,n,dh) bf16, n padded to NPAD_)
//   1: scatter into v^T buffer ((b,h,dv,key) bf16, key padded to KPAD_)
//   2: xio[row*N+col] += val   (f32 residual accumulate)
//   3: xio[row*N+col]  = val   (f32 store)
template <int EPI>
__global__ __launch_bounds__(256)
void gemm_bf16(const unsigned short* __restrict__ A, int lda,
               const unsigned short* __restrict__ BT,
               const float* __restrict__ bias, int K, int N,
               float* __restrict__ xio, unsigned short* __restrict__ out0,
               unsigned short* __restrict__ out1) {
  __shared__ unsigned short Alds[2][128 * 40];  // 128 rows x 32 halves, stride 40
  __shared__ unsigned short Blds[2][128 * 40];  // 128 cols (as rows) x 32 halves

  const int wave = threadIdx.x >> 5;
  const int wm = wave & 3, wn = wave >> 2;      // 4 x 2 wave grid
  const int lane = threadIdx.x & 31;
  const int g = lane >> 4, nn = lane & 15;
  const int m0 = blockIdx.x * 128;
  const int n0 = blockIdx.y * 128;

  // per-thread staging addresses (two rows, 16B segment each)
  const int r0 = threadIdx.x >> 2, seg = threadIdx.x & 3;
  const unsigned short* Ar0 = A + (size_t)(m0 + r0) * lda + seg * 8;
  const unsigned short* Ar1 = A + (size_t)(m0 + r0 + 64) * lda + seg * 8;
  const unsigned short* Br0 = BT + (size_t)(n0 + r0) * K + seg * 8;
  const unsigned short* Br1 = BT + (size_t)(n0 + r0 + 64) * K + seg * 8;

  v8f acc[2][4];
#pragma unroll
  for (int s = 0; s < 2; ++s)
#pragma unroll
    for (int t = 0; t < 4; ++t) acc[s][t] = v8f{};

  uint4 ra0, ra1, rb0, rb1;
  // preload tile 0 into buffer 0
  ra0 = *(const uint4*)(Ar0);
  ra1 = *(const uint4*)(Ar1);
  rb0 = *(const uint4*)(Br0);
  rb1 = *(const uint4*)(Br1);
  *(uint4*)(&Alds[0][r0 * 40 + seg * 8]) = ra0;
  *(uint4*)(&Alds[0][(r0 + 64) * 40 + seg * 8]) = ra1;
  *(uint4*)(&Blds[0][r0 * 40 + seg * 8]) = rb0;
  *(uint4*)(&Blds[0][(r0 + 64) * 40 + seg * 8]) = rb1;

  int p = 0;
  for (int k0 = 0; k0 < K; k0 += 32) {
    __syncthreads();
    const bool more = (k0 + 32) < K;
    if (more) {  // issue global loads for next tile (overlap with WMMAs below)
      ra0 = *(const uint4*)(Ar0 + k0 + 32);
      ra1 = *(const uint4*)(Ar1 + k0 + 32);
      rb0 = *(const uint4*)(Br0 + k0 + 32);
      rb1 = *(const uint4*)(Br1 + k0 + 32);
    }
    if ((k0 + 64) < K) {  // pull tile k+2 toward the WGP
      __builtin_prefetch(Ar0 + k0 + 64, 0, 1);
      __builtin_prefetch(Br0 + k0 + 64, 0, 1);
    }

    v16bf af[2], bfg[4];
#pragma unroll
    for (int s = 0; s < 2; ++s)
      af[s] = load_frag_a(&Alds[p][(wm * 32 + s * 16) * 40], 40, 15);
#pragma unroll
    for (int t = 0; t < 4; ++t)
      bfg[t] = load_frag_b(&Blds[p][(wn * 64 + t * 16) * 40], 40, 15);
#pragma unroll
    for (int s = 0; s < 2; ++s)
#pragma unroll
      for (int t = 0; t < 4; ++t) acc[s][t] = wmma_bf16(af[s], bfg[t], acc[s][t]);

    if (more) {  // stage next tile into the other buffer
      *(uint4*)(&Alds[p ^ 1][r0 * 40 + seg * 8]) = ra0;
      *(uint4*)(&Alds[p ^ 1][(r0 + 64) * 40 + seg * 8]) = ra1;
      *(uint4*)(&Blds[p ^ 1][r0 * 40 + seg * 8]) = rb0;
      *(uint4*)(&Blds[p ^ 1][(r0 + 64) * 40 + seg * 8]) = rb1;
    }
    p ^= 1;
  }

  // epilogue (C layout: element e of lane 16g+nn -> row e+8g, col nn)
#pragma unroll
  for (int s = 0; s < 2; ++s)
#pragma unroll
    for (int t = 0; t < 4; ++t) {
#pragma unroll
      for (int e = 0; e < 8; ++e) {
        int row = m0 + wm * 32 + s * 16 + e + 8 * g;
        int col = n0 + wn * 64 + t * 16 + nn;
        float val = acc[s][t][e] + bias[col];
        if (EPI == 0) {
          int bb = row / N_, ntok = row % N_;
          if (col < INNER_) {
            int hh = col >> 6, dh = col & 63;
            out0[((size_t)(bb * HEADS_ + hh) * NPAD_ + ntok) * DH_ + dh] = f2bf(val);
          } else {
            int c2 = col - INNER_;
            int hh = c2 >> 6, dh = c2 & 63;
            out1[((size_t)(bb * HEADS_ + hh) * NPAD_ + ntok) * DH_ + dh] = f2bf(val);
          }
        } else if (EPI == 1) {
          int bb = row / N_, ntok = row % N_;
          int hh = col / DV_, dv = col % DV_;
          out0[((size_t)(bb * HEADS_ + hh) * DV_ + dv) * KPAD_ + ntok] = f2bf(val);
        } else if (EPI == 2) {
          xio[(size_t)row * N + col] += val;
        } else {
          xio[(size_t)row * N + col] = val;
        }
      }
    }
}

// ---------------- flash attention (one (b,h) per block) ----------------
__global__ __launch_bounds__(256)
void attn_kernel(const unsigned short* __restrict__ qb,
                 const unsigned short* __restrict__ kb,
                 const unsigned short* __restrict__ vt,
                 unsigned short* __restrict__ ob) {
  __shared__ unsigned short pstage[8][16 * 40];  // per-wave prob staging 16x32
  const int bh = blockIdx.x;
  const int b = bh >> 3, h = bh & 7;
  const int wave = threadIdx.x >> 5;
  const int lane = threadIdx.x & 31;
  const int g = lane >> 4, nn = lane & 15;

  for (int t = wave; t < 13; t += 8) {          // q tiles of 16 rows
    const unsigned short* qbase = qb + ((size_t)bh * NPAD_ + t * 16) * DH_;
    v16bf qf0 = load_frag_a(qbase, DH_, 15);       // dh 0..31
    v16bf qf1 = load_frag_a(qbase + 32, DH_, 15);  // dh 32..63

    float mrow[8], lrow[8];
    v8f oacc[7];
#pragma unroll
    for (int e = 0; e < 8; ++e) { mrow[e] = -3.0e38f; lrow[e] = 0.f; }
#pragma unroll
    for (int d = 0; d < 7; ++d) oacc[d] = v8f{};

    for (int kp = 0; kp < 7; ++kp) {            // 32 keys per step
      int kt0 = 2 * kp, kt1 = 2 * kp + 1;
      int kr0 = kt0 * 16, rc0 = 195 - kr0;
      int kr1 = kt1 * 16, rc1 = 195 - kr1;
      if (rc1 < 0) { kr1 = 192; rc1 = 3; }      // fully-masked pad tile

      v8f s0 = v8f{}, s1 = v8f{};
      {
        const unsigned short* kb0 = kb + ((size_t)bh * NPAD_ + kr0) * DH_;
        s0 = wmma_bf16(qf0, load_frag_b(kb0, DH_, rc0), s0);
        s0 = wmma_bf16(qf1, load_frag_b(kb0 + 32, DH_, rc0), s0);
        const unsigned short* kb1 = kb + ((size_t)bh * NPAD_ + kr1) * DH_;
        s1 = wmma_bf16(qf0, load_frag_b(kb1, DH_, rc1), s1);
        s1 = wmma_bf16(qf1, load_frag_b(kb1 + 32, DH_, rc1), s1);
      }

      // scale + mask padded keys
      int key0 = kp * 32 + nn, key1 = kp * 32 + 16 + nn;
#pragma unroll
      for (int e = 0; e < 8; ++e) {
        float a0 = s0[e] * 0.125f;              // DH^-0.5
        float a1 = s1[e] * 0.125f;
        if (key0 >= N_) a0 = -1e30f;
        if (key1 >= N_) a1 = -1e30f;
        s0[e] = a0; s1[e] = a1;
      }

      // online softmax update (row reductions across the 16-lane half)
      float alpha[8];
#pragma unroll
      for (int e = 0; e < 8; ++e) {
        float v = fmaxf(s0[e], s1[e]);
#pragma unroll
        for (int off = 1; off < 16; off <<= 1) v = fmaxf(v, __shfl_xor(v, off, 32));
        float mn = fmaxf(mrow[e], v);
        alpha[e] = __expf(mrow[e] - mn);
        mrow[e] = mn;
      }
#pragma unroll
      for (int e = 0; e < 8; ++e) {
        float p0 = __expf(s0[e] - mrow[e]);
        float p1 = __expf(s1[e] - mrow[e]);
        s0[e] = p0; s1[e] = p1;
        float v = p0 + p1;
#pragma unroll
        for (int off = 1; off < 16; off <<= 1) v += __shfl_xor(v, off, 32);
        lrow[e] = lrow[e] * alpha[e] + v;
      }

      // stage probs (row-major 16x32 bf16), re-fragment as WMMA A operand
#pragma unroll
      for (int e = 0; e < 8; ++e) {
        pstage[wave][(e + 8 * g) * 40 + nn] = f2bf(s0[e]);
        pstage[wave][(e + 8 * g) * 40 + 16 + nn] = f2bf(s1[e]);
      }
      asm volatile("s_wait_dscnt 0x0" ::: "memory");  // wave-local LDS RAW
      v16bf pf = load_frag_a(&pstage[wave][0], 40, 15);

#pragma unroll
      for (int d = 0; d < 7; ++d)
#pragma unroll
        for (int e = 0; e < 8; ++e) oacc[d][e] *= alpha[e];

#pragma unroll
      for (int d = 0; d < 7; ++d) {
        const unsigned short* vb = vt + ((size_t)bh * DV_ + d * 16) * KPAD_ + kp * 32;
        oacc[d] = wmma_bf16(pf, load_frag_b(vb, KPAD_, 15), oacc[d]);
      }
    }

    float inv[8];
#pragma unroll
    for (int e = 0; e < 8; ++e) inv[e] = 1.0f / lrow[e];
#pragma unroll
    for (int d = 0; d < 7; ++d)
#pragma unroll
      for (int e = 0; e < 8; ++e) {
        int qrow = t * 16 + e + 8 * g;
        if (qrow < N_)
          ob[((size_t)(b * N_ + qrow)) * VD_ + h * DV_ + d * 16 + nn] =
              f2bf(oacc[d][e] * inv[e]);
      }
  }
}

// ---------------- host side ----------------
extern "C" void kernel_launch(void* const* d_in, const int* in_sizes, int n_in,
                              void* d_out, int out_size, void* d_ws, size_t ws_size,
                              hipStream_t stream) {
  (void)in_sizes; (void)n_in; (void)out_size; (void)ws_size;
  const float* x_in  = (const float*)d_in[0];
  const float* bn1_g = (const float*)d_in[1];
  const float* bn1_b = (const float*)d_in[2];
  const float* Wqk   = (const float*)d_in[3];
  const float* bqk   = (const float*)d_in[4];
  const float* Wv    = (const float*)d_in[5];
  const float* bv    = (const float*)d_in[6];
  const float* Wo    = (const float*)d_in[7];
  const float* bo    = (const float*)d_in[8];
  const float* bn2_g = (const float*)d_in[9];
  const float* bn2_b = (const float*)d_in[10];
  const float* W1    = (const float*)d_in[11];
  const float* b1    = (const float*)d_in[12];
  const float* bn3_g = (const float*)d_in[13];
  const float* bn3_b = (const float*)d_in[14];
  const float* W2    = (const float*)d_in[15];
  const float* b2    = (const float*)d_in[16];

  char* ws = (char*)d_ws;
  size_t off = 0;
  auto give = [&](size_t bytes) -> char* {
    char* p = ws + off;
    off += (bytes + 255) & ~(size_t)255;
    return p;
  };

  unsigned short* hb   = (unsigned short*)give((size_t)MROWS_ * DIM_ * 2);
  unsigned short* qbuf = (unsigned short*)give((size_t)B_ * HEADS_ * NPAD_ * DH_ * 2);
  unsigned short* kbuf = (unsigned short*)give((size_t)B_ * HEADS_ * NPAD_ * DH_ * 2);
  unsigned short* vtb  = (unsigned short*)give((size_t)B_ * HEADS_ * DV_ * KPAD_ * 2);
  unsigned short* obuf = (unsigned short*)give((size_t)MROWS_ * VD_ * 2);
  float*          fbuf = (float*)give((size_t)MROWS_ * DIM_ * 4);
  unsigned short* gbuf = (unsigned short*)give((size_t)MROWS_ * DIM_ * 2);
  float*          scl  = (float*)give(DIM_ * 4);
  float*          shf  = (float*)give(DIM_ * 4);
  unsigned short* wqkT = (unsigned short*)give((size_t)6 * 1024 * 384 * 2);
  unsigned short* wvT  = (unsigned short*)give((size_t)6 * VD_ * DIM_ * 2);
  unsigned short* woT  = (unsigned short*)give((size_t)6 * DIM_ * VD_ * 2);
  unsigned short* w1T  = (unsigned short*)give((size_t)6 * DIM_ * DIM_ * 2);
  unsigned short* w2T  = (unsigned short*)give((size_t)6 * DIM_ * DIM_ * 2);

  // weights -> transposed bf16 (redone every call; deterministic)
  for (int i = 0; i < 6; ++i) {
    wtrans_kernel<<<(384 * 1024 + 255) / 256, 256, 0, stream>>>(
        Wqk + (size_t)i * 384 * 1024, wqkT + (size_t)i * 1024 * 384, 384, 1024);
    wtrans_kernel<<<(DIM_ * VD_ + 255) / 256, 256, 0, stream>>>(
        Wv + (size_t)i * DIM_ * VD_, wvT + (size_t)i * VD_ * DIM_, DIM_, VD_);
    wtrans_kernel<<<(VD_ * DIM_ + 255) / 256, 256, 0, stream>>>(
        Wo + (size_t)i * VD_ * DIM_, woT + (size_t)i * DIM_ * VD_, VD_, DIM_);
    wtrans_kernel<<<(DIM_ * DIM_ + 255) / 256, 256, 0, stream>>>(
        W1 + (size_t)i * DIM_ * DIM_, w1T + (size_t)i * DIM_ * DIM_, DIM_, DIM_);
    wtrans_kernel<<<(DIM_ * DIM_ + 255) / 256, 256, 0, stream>>>(
        W2 + (size_t)i * DIM_ * DIM_, w2T + (size_t)i * DIM_ * DIM_, DIM_, DIM_);
  }

  // running activation x lives in d_out (f32)
  hipMemcpyAsync(d_out, (const void*)x_in, (size_t)MROWS_ * DIM_ * 4,
                 hipMemcpyDeviceToDevice, stream);
  float* xb = (float*)d_out;

  const int EW_GRID = (MROWS_ * DIM_ + 255) / 256;

  for (int i = 0; i < 6; ++i) {
    // --- attention branch ---
    bn_stats_kernel<<<DIM_, 256, 0, stream>>>(xb, bn1_g + i * DIM_, bn1_b + i * DIM_,
                                              scl, shf, MROWS_, DIM_);
    norm_kernel<<<EW_GRID, 256, 0, stream>>>(xb, scl, shf, hb, MROWS_ * DIM_, DIM_);

    gemm_bf16<0><<<dim3(98, 8), 256, 0, stream>>>(
        hb + EMBED_, DIM_, wqkT + (size_t)i * 1024 * 384, bqk + (size_t)i * 1024,
        CDIM_, 2 * INNER_, nullptr, qbuf, kbuf);
    gemm_bf16<1><<<dim3(98, 7), 256, 0, stream>>>(
        hb, DIM_, wvT + (size_t)i * VD_ * DIM_, bv + (size_t)i * VD_,
        DIM_, VD_, nullptr, vtb, nullptr);

    attn_kernel<<<B_ * HEADS_, 256, 0, stream>>>(qbuf, kbuf, vtb, obuf);

    gemm_bf16<2><<<dim3(98, 6), 256, 0, stream>>>(
        obuf, VD_, woT + (size_t)i * DIM_ * VD_, bo + (size_t)i * DIM_,
        VD_, DIM_, xb, nullptr, nullptr);

    // --- FFN branch ---
    bn_stats_kernel<<<DIM_, 256, 0, stream>>>(xb, bn2_g + i * DIM_, bn2_b + i * DIM_,
                                              scl, shf, MROWS_, DIM_);
    norm_kernel<<<EW_GRID, 256, 0, stream>>>(xb, scl, shf, hb, MROWS_ * DIM_, DIM_);

    gemm_bf16<3><<<dim3(98, 6), 256, 0, stream>>>(
        hb, DIM_, w1T + (size_t)i * DIM_ * DIM_, b1 + (size_t)i * DIM_,
        DIM_, DIM_, fbuf, nullptr, nullptr);

    bn_stats_kernel<<<DIM_, 256, 0, stream>>>(fbuf, bn3_g + i * DIM_, bn3_b + i * DIM_,
                                              scl, shf, MROWS_, DIM_);
    gelu_kernel<<<EW_GRID, 256, 0, stream>>>(fbuf, scl, shf, gbuf, MROWS_ * DIM_, DIM_);

    gemm_bf16<2><<<dim3(98, 6), 256, 0, stream>>>(
        gbuf, DIM_, w2T + (size_t)i * DIM_ * DIM_, b2 + (size_t)i * DIM_,
        DIM_, DIM_, xb, nullptr, nullptr);
  }
}
